// NSFFM_85255100825767
// MI455X (gfx1250) — matrix operand order repaired and checked by reading, not compiled, not run on hardware
//
#include <hip/hip_runtime.h>
#include <hip/hip_bf16.h>
#include <stdint.h>

// ---------------------------------------------------------------------------
// NSFFM on MI455X (gfx1250, wave32).
// GEMMs in bf16 with f32 accumulation via V_WMMA_F32_16X16X32_BF16.
// Scan normalization factorized: sum_{r,c}|tr_r*ct_c| = (sum|tr|)(sum|ct|).
// ---------------------------------------------------------------------------

typedef __bf16 bf16_t;
typedef __attribute__((ext_vector_type(16))) __bf16 v16bf;
typedef __attribute__((ext_vector_type(8)))  float  v8f;

#define T_LEN 4096
#define D_DIM 512
#define TR_N  64
#define CC_N  64
#define F_DIM 8192   // 2*TR*CC

__device__ __forceinline__ bf16_t f2bf(float f) { return (bf16_t)f; }

// ---------------- elementwise helpers ----------------
__global__ void cvt_bf16_kernel(const float* __restrict__ src,
                                bf16_t* __restrict__ dst, int n) {
  int i = blockIdx.x * blockDim.x + threadIdx.x;
  if (i < n) dst[i] = f2bf(src[i]);
}

__global__ void init_x_kernel(const float* __restrict__ x,
                              float* __restrict__ xcur,
                              bf16_t* __restrict__ xb, int n) {
  int i = blockIdx.x * blockDim.x + threadIdx.x;
  if (i < n) { float v = x[i]; xcur[i] = v; xb[i] = f2bf(v); }
}

__global__ void residual_kernel(float* __restrict__ xcur,
                                const float* __restrict__ z,
                                bf16_t* __restrict__ xb, int n) {
  int i = blockIdx.x * blockDim.x + threadIdx.x;
  if (i < n) { float v = xcur[i] + z[i]; xcur[i] = v; xb[i] = f2bf(v); }
}

// ---------------- bf16 WMMA GEMM: C[M,N] = A[M,K] * B[N,K]^T ----------------
// Block: 256 threads = 8 waves (4 in M x 2 in N). Block tile 128x64.
// Wave tile 32x32 = 2x2 fragments of v_wmma_f32_16x16x32_bf16.
// Requires M%128==0, N%64==0, K%32==0 (guaranteed by caller).
union FragBF { v16bf v; unsigned int u[8]; };

__global__ __launch_bounds__(256) void gemm_bf16_wmma(
    const bf16_t* __restrict__ A, const bf16_t* __restrict__ B,
    float* __restrict__ C, int M, int N, int K) {
  __shared__ __align__(16) bf16_t As[128 * 32];
  __shared__ __align__(16) bf16_t Bs[64 * 32];
  const unsigned int* As_u = (const unsigned int*)As;
  const unsigned int* Bs_u = (const unsigned int*)Bs;

  const int tid  = threadIdx.x;
  const int lane = tid & 31;
  const int w    = tid >> 5;
  const int wm   = w >> 1;     // 0..3
  const int wn   = w & 1;      // 0..1
  const int bm   = blockIdx.x * 128;
  const int bn   = blockIdx.y * 64;
  const int hi   = lane >> 4;  // lane group
  const int lo   = lane & 15;

  v8f acc[2][2] = {};

  for (int k0 = 0; k0 < K; k0 += 32) {
    __syncthreads();
    // A tile: 128x32 bf16 = 512 uint4 -> 2 per thread
    #pragma unroll
    for (int i = 0; i < 2; i++) {
      int idx = i * 256 + tid;
      int r = idx >> 2, c4 = idx & 3;
      *(uint4*)(&As[r * 32 + c4 * 8]) =
          *(const uint4*)(A + (size_t)(bm + r) * K + k0 + c4 * 8);
    }
    // B tile: 64x32 bf16 = 256 uint4 -> 1 per thread
    {
      int r = tid >> 2, c4 = tid & 3;
      *(uint4*)(&Bs[r * 32 + c4 * 8]) =
          *(const uint4*)(B + (size_t)(bn + r) * K + k0 + c4 * 8);
    }
    __syncthreads();

    // Gather fragments per documented 16-bit A (16x32) / B (32x16) layouts.
    FragBF af[2], bfr[2];
    #pragma unroll
    for (int s = 0; s < 2; s++) {
      int m = wm * 32 + s * 16 + lo;
      #pragma unroll
      for (int v = 0; v < 8; v++) {
        int k = ((v & 4) << 2) + (hi << 3) + ((v & 3) << 1);  // paired K
        af[s].u[v] = As_u[m * 16 + (k >> 1)];
      }
      int n = wn * 32 + s * 16 + lo;
      #pragma unroll
      for (int v = 0; v < 8; v++) {
        int k = ((v & 4) << 2) + (hi << 3) + ((v & 3) << 1);
        bfr[s].u[v] = Bs_u[n * 16 + (k >> 1)];
      }
    }
    #pragma unroll
    for (int i = 0; i < 2; i++)
      #pragma unroll
      for (int j = 0; j < 2; j++)
        acc[i][j] = __builtin_amdgcn_wmma_f32_16x16x32_bf16(
            false, af[i].v, false, bfr[j].v, (short)0, acc[i][j], false, false);
  }

  // C/D layout: VGPR r -> M = r + 8*hi, N = lo.
  #pragma unroll
  for (int i = 0; i < 2; i++)
    #pragma unroll
    for (int j = 0; j < 2; j++) {
      int m0 = bm + wm * 32 + i * 16 + 8 * hi;
      int n0 = bn + wn * 32 + j * 16 + lo;
      #pragma unroll
      for (int r = 0; r < 8; r++)
        C[(size_t)(m0 + r) * N + n0] = acc[i][j][r];
    }
}

// ---------------- denom[t] = 1e-8 + (sum_r |tr|)(sum_c |ct|) ----------------
__global__ __launch_bounds__(256) void denom_kernel(
    const float* __restrict__ tr, const float* __restrict__ ct,
    float* __restrict__ denom) {
  int lane = threadIdx.x & 31;
  int t = blockIdx.x * 8 + (threadIdx.x >> 5);
  float s1 = fabsf(tr[t * 64 + lane]) + fabsf(tr[t * 64 + lane + 32]);
  float s2 = fabsf(ct[t * 64 + lane]) + fabsf(ct[t * 64 + lane + 32]);
  #pragma unroll
  for (int m = 16; m > 0; m >>= 1) {
    s1 += __shfl_xor(s1, m, 32);
    s2 += __shfl_xor(s2, m, 32);
  }
  if (lane == 0) denom[t] = 1e-8f + s1 * s2;
}

// ---------------- sequential complex scan + log-polar features ----------------
// block = 64 threads (c), grid = 64 blocks (r). dt == 1 between steps.
__global__ __launch_bounds__(64) void scan_kernel(
    const float* __restrict__ tr, const float* __restrict__ ct,
    const float* __restrict__ denom, const unsigned char* __restrict__ start,
    const float* __restrict__ state0, const float* __restrict__ a,
    const float* __restrict__ bfreq, bf16_t* __restrict__ scaled) {
  int r = blockIdx.x;
  int c = threadIdx.x;
  float dr = expf(-fabsf(a[r]));
  float sb, cb;
  sincosf(bfreq[c], &sb, &cb);
  float dre = dr * cb, dim = dr * sb;
  float sr = state0[r * 64 + c], si = 0.f;
  int idx = r * 64 + c;
  for (int t = 0; t < T_LEN; t++) {
    float trv = tr[t * 64 + r];
    float ctv = ct[t * 64 + c];
    float pre = fabsf(trv * ctv) / denom[t];
    if (start[t]) { sr = 0.f; si = 0.f; }
    float nsr = sr * dre - si * dim + pre;
    float nsi = sr * dim + si * dre;
    sr = nsr; si = nsi;
    float av  = sqrtf(sr * sr + si * si);
    float mag = log1pf(av);
    float inv = (av > 0.f) ? (1.f / av) : 0.f;
    float snp = si * inv;
    float csp = (av > 0.f) ? sr * inv : 1.f;  // angle(0)=0 -> cos=1 (mag=0 anyway)
    scaled[(size_t)t * F_DIM + idx]        = f2bf(mag * snp);
    scaled[(size_t)t * F_DIM + 4096 + idx] = f2bf(mag * csp);
  }
}

// ---------------- fused bias + LayerNorm + leaky_relu ----------------
// One wave per row (D=512 -> 16 elems/lane). outF / outB optional.
__global__ __launch_bounds__(256) void ln_lrelu_kernel(
    const float* __restrict__ in, const float* __restrict__ bias,
    const float* __restrict__ gamma, const float* __restrict__ beta,
    float* __restrict__ outF, bf16_t* __restrict__ outB) {
  int lane = threadIdx.x & 31;
  int row = blockIdx.x * 8 + (threadIdx.x >> 5);
  const float* p = in + (size_t)row * D_DIM;
  float v[16];
  float s = 0.f;
  #pragma unroll
  for (int j = 0; j < 16; j++) {
    int d = j * 32 + lane;
    v[j] = p[d] + bias[d];
    s += v[j];
  }
  #pragma unroll
  for (int m = 16; m > 0; m >>= 1) s += __shfl_xor(s, m, 32);
  float mean = s * (1.f / 512.f);
  float q = 0.f;
  #pragma unroll
  for (int j = 0; j < 16; j++) { float d0 = v[j] - mean; q += d0 * d0; }
  #pragma unroll
  for (int m = 16; m > 0; m >>= 1) q += __shfl_xor(q, m, 32);
  float rstd = 1.f / sqrtf(q * (1.f / 512.f) + 1e-5f);
  #pragma unroll
  for (int j = 0; j < 16; j++) {
    int d = j * 32 + lane;
    float y = (v[j] - mean) * rstd * gamma[d] + beta[d];
    y = (y >= 0.f) ? y : 0.01f * y;
    size_t o = (size_t)row * D_DIM + d;
    if (outF) outF[o] = y;
    if (outB) outB[o] = f2bf(y);
  }
}

// ---------------------------------------------------------------------------
extern "C" void kernel_launch(void* const* d_in, const int* in_sizes, int n_in,
                              void* d_out, int out_size, void* d_ws, size_t ws_size,
                              hipStream_t stream) {
  (void)in_sizes; (void)n_in; (void)out_size; (void)ws_size;
  const float* x     = (const float*)d_in[0];
  const float* state = (const float*)d_in[1];
  const unsigned char* start = (const unsigned char*)d_in[2];  // bool array
  const float* Wtr = (const float*)d_in[3];
  const float* Wc  = (const float*)d_in[4];
  const float* a   = (const float*)d_in[5];
  const float* bfq = (const float*)d_in[6];
  const float* W0  = (const float*)d_in[7];
  const float* b0  = (const float*)d_in[8];
  const float* g0  = (const float*)d_in[9];
  const float* be0 = (const float*)d_in[10];
  const float* W1  = (const float*)d_in[11];
  const float* b1  = (const float*)d_in[12];
  const float* g1  = (const float*)d_in[13];
  const float* be1 = (const float*)d_in[14];

  const int T = T_LEN, D = D_DIM, F = F_DIM;

  char* ws = (char*)d_ws;
  size_t off = 0;
  auto carve = [&](size_t bytes) -> char* {
    char* p = ws + off;
    off += (bytes + 255) & ~((size_t)255);
    return p;
  };
  float*  xcur    = (float*) carve((size_t)T * D * 4);
  bf16_t* xb      = (bf16_t*)carve((size_t)T * D * 2);
  bf16_t* wtrb    = (bf16_t*)carve((size_t)TR_N * D * 2);
  bf16_t* wcb     = (bf16_t*)carve((size_t)CC_N * D * 2);
  bf16_t* w0b     = (bf16_t*)carve((size_t)D * F * 2);
  bf16_t* w1b     = (bf16_t*)carve((size_t)D * D * 2);
  float*  tr      = (float*) carve((size_t)T * TR_N * 4);
  float*  ct      = (float*) carve((size_t)T * CC_N * 4);
  float*  denom   = (float*) carve((size_t)T * 4);
  bf16_t* scaledb = (bf16_t*)carve((size_t)T * F * 2);
  float*  zc      = (float*) carve((size_t)T * D * 4);
  bf16_t* z0b     = (bf16_t*)carve((size_t)T * D * 2);
  float*  z1      = (float*) carve((size_t)T * D * 4);

  const int n_xd = T * D;
  init_x_kernel<<<dim3(n_xd / 256), dim3(256), 0, stream>>>(x, xcur, xb, n_xd);

  for (int l = 0; l < 3; l++) {
    // convert this layer's weights to bf16
    cvt_bf16_kernel<<<dim3((TR_N * D) / 256), dim3(256), 0, stream>>>(
        Wtr + (size_t)l * TR_N * D, wtrb, TR_N * D);
    cvt_bf16_kernel<<<dim3((CC_N * D) / 256), dim3(256), 0, stream>>>(
        Wc + (size_t)l * CC_N * D, wcb, CC_N * D);
    cvt_bf16_kernel<<<dim3((D * F) / 256), dim3(256), 0, stream>>>(
        W0 + (size_t)l * D * F, w0b, D * F);
    cvt_bf16_kernel<<<dim3((D * D) / 256), dim3(256), 0, stream>>>(
        W1 + (size_t)l * D * D, w1b, D * D);

    // tr = x @ Wtr^T , ct = x @ Wc^T   (M=4096, N=64, K=512)
    gemm_bf16_wmma<<<dim3(T / 128, TR_N / 64), dim3(256), 0, stream>>>(
        xb, wtrb, tr, T, TR_N, D);
    gemm_bf16_wmma<<<dim3(T / 128, CC_N / 64), dim3(256), 0, stream>>>(
        xb, wcb, ct, T, CC_N, D);

    // separable normalization denominator
    denom_kernel<<<dim3(T / 8), dim3(256), 0, stream>>>(tr, ct, denom);

    // sequential oscillator scan -> bf16 'scaled' features (T x 8192)
    scan_kernel<<<dim3(64), dim3(64), 0, stream>>>(
        tr, ct, denom, start, state + (size_t)l * TR_N * CC_N,
        a + (size_t)l * TR_N, bfq + (size_t)l * CC_N, scaledb);

    // z0 = leaky(LN(scaled @ W0^T + b0))   (M=4096, N=512, K=8192)
    gemm_bf16_wmma<<<dim3(T / 128, D / 64), dim3(256), 0, stream>>>(
        scaledb, w0b, zc, T, D, F);
    ln_lrelu_kernel<<<dim3(T / 8), dim3(256), 0, stream>>>(
        zc, b0 + (size_t)l * D, g0 + (size_t)l * D, be0 + (size_t)l * D,
        nullptr, z0b);

    // z = leaky(LN(z0 @ W1^T + b1))        (M=4096, N=512, K=512)
    gemm_bf16_wmma<<<dim3(T / 128, D / 64), dim3(256), 0, stream>>>(
        z0b, w1b, zc, T, D, D);
    float* zOut = (l == 2) ? (float*)d_out : z1;
    ln_lrelu_kernel<<<dim3(T / 8), dim3(256), 0, stream>>>(
        zc, b1 + (size_t)l * D, g1 + (size_t)l * D, be1 + (size_t)l * D,
        zOut, nullptr);

    // x = x + z for next layer
    if (l < 2)
      residual_kernel<<<dim3(n_xd / 256), dim3(256), 0, stream>>>(
          xcur, z1, xb, n_xd);
  }
}
